// TreeTransformerSparsePostv2_10196252361128
// MI455X (gfx1250) — matrix operand order repaired and checked by reading, not compiled, not run on hardware
//
#include <hip/hip_runtime.h>

// ---------------- model dims (fixed by reference) ----------------
#define B_SZ   256
#define LQ     32
#define LP     32
#define SUBW   8
#define DEPTH  8
#define RREG   48
#define HID    512
#define NHEAD  8
#define DHEAD  64
#define FFDIM  2048
#define ANSV   1853
#define EMBD   300

typedef __attribute__((ext_vector_type(16))) __bf16 v16bf;
typedef __attribute__((ext_vector_type(8)))  float  v8f;
typedef unsigned int u32x4 __attribute__((ext_vector_type(4)));
typedef int          i32x8 __attribute__((ext_vector_type(8)));
typedef int          i32x4 __attribute__((ext_vector_type(4)));

union Frag { v16bf v; unsigned int u[8]; };
static_assert(sizeof(Frag) == 32, "frag size");

#if __has_builtin(__builtin_amdgcn_tensor_load_to_lds) && __has_builtin(__builtin_amdgcn_s_wait_tensorcnt)
#define GEMM_USE_TDM 1
#else
#define GEMM_USE_TDM 0
#endif

__device__ inline unsigned short f2bf(float f) {
  unsigned int u = __float_as_uint(f);
  u += 0x7FFFu + ((u >> 16) & 1u);       // round-to-nearest-even
  return (unsigned short)(u >> 16);
}

__device__ inline unsigned lds_byte_offset(const void* p) {
  // generic -> LDS(as3) addrspacecast, then ptrtoint: 32-bit LDS byte address
  return (unsigned)(unsigned long long)(__attribute__((address_space(3))) const void*)p;
}

// ---------------- f32 -> bf16 weight pre-conversion ----------------
__global__ void cvt_f32_bf16_kernel(const float* __restrict__ src,
                                    unsigned short* __restrict__ dst, int n) {
  int i = blockIdx.x * 256 + threadIdx.x;
  if (i < n) dst[i] = f2bf(src[i]);
}

// ---------------- generic WMMA GEMM: C = relu?(A[M,K]_f32 x W[K,N]_bf16 + bias) ----------------
// A staged f32->bf16 into LDS [m][k]; W tile DMA'd (TDM) into LDS [k][n] bf16,
// double-buffered so tile s+1 transfers while tile s runs on the WMMA pipe.
#define BM 128
#define BN 128
#define BK 32

template <bool RELU>
__global__ __launch_bounds__(256) void gemm_bias_kernel(
    const float* __restrict__ A, const unsigned short* __restrict__ W,
    const float* __restrict__ bias, float* __restrict__ C,
    int M, int N, int K)
{
  __shared__ __align__(16) unsigned short sA[2][BM * BK];   // [m][k] bf16
  __shared__ __align__(16) unsigned short sB[2][BK * BN];   // [k][n] bf16
  const int tid  = threadIdx.x;
  const int lane = tid & 31;
  const int wave = tid >> 5;
  const int wm = wave & 3;       // 4 row-groups of 32
  const int wn = wave >> 2;      // 2 col-groups of 64
  const int bm0 = blockIdx.y * BM;
  const int bn0 = blockIdx.x * BN;
  const int nsteps = (K + BK - 1) / BK;

  v8f acc[2][4];
#pragma unroll
  for (int i = 0; i < 2; ++i)
#pragma unroll
    for (int j = 0; j < 4; ++j)
#pragma unroll
      for (int r = 0; r < 8; ++r) acc[i][j][r] = 0.0f;

  const int lhalf = (lane < 16) ? 0 : 1;
  const int mlane = lane & 15;

#if GEMM_USE_TDM
  // D# group1 is K-step invariant: bf16 tensor [K, N], tile [BK, BN]
  i32x8 g1;
  g1[0] = 0x00010000;                                            // wgmask=0, data_size=1 (2B)
  g1[1] = (int)(((unsigned)N & 0xffffu) << 16);                  // tensor_dim0 lo16
  g1[2] = (int)(((unsigned)N >> 16) | (((unsigned)K & 0xffffu) << 16)); // dim0 hi | dim1 lo
  g1[3] = (int)(((unsigned)K >> 16) | ((unsigned)BN << 16));     // dim1 hi | tile_dim0
  g1[4] = BK;                                                    // tile_dim1 | tile_dim2=0
  g1[5] = N;                                                     // tensor_dim0_stride lo32
  g1[6] = 0;
  g1[7] = 0;
  i32x4 gz;
  gz[0] = 0; gz[1] = 0; gz[2] = 0; gz[3] = 0;
  i32x8 gz8;
#pragma unroll
  for (int i = 0; i < 8; ++i) gz8[i] = 0;
#endif

  // ---- prologue: tile 0 ----
  {
#if GEMM_USE_TDM
    if (wave == 0) {
      unsigned long long ga = (unsigned long long)(const void*)(W + (size_t)bn0);
      u32x4 g0;
      g0[0] = 1u;                                   // count=1, user descriptor
      g0[1] = lds_byte_offset(&sB[0][0]);
      g0[2] = (unsigned)ga;
      g0[3] = ((unsigned)(ga >> 32) & 0x01FFFFFFu) | 0x80000000u;  // addr[56:32] | type=2
      __builtin_amdgcn_tensor_load_to_lds(g0, g1, gz, gz, gz8, 0);
    }
#else
    for (int i = tid; i < BK * BN; i += 256) {
      int kk = i >> 7, n = i & (BN - 1);
      int gn = bn0 + n;
      sB[0][i] = (kk < K && gn < N) ? W[(size_t)kk * N + gn] : (unsigned short)0;
    }
#endif
    for (int i = tid; i < BM * BK; i += 256) {
      int m = i >> 5, kk = i & 31;
      int gm = bm0 + m;
      float v = (gm < M && kk < K) ? A[(size_t)gm * K + kk] : 0.0f;
      sA[0][m * BK + kk] = f2bf(v);
    }
#if GEMM_USE_TDM
    if (wave == 0) __builtin_amdgcn_s_wait_tensorcnt(0);
#endif
    __syncthreads();
  }

  for (int s = 0; s < nsteps; ++s) {
    const int buf = s & 1;
    const int k0n = (s + 1) * BK;
    if (s + 1 < nsteps) {
      // ---- launch transfers for tile s+1 into the other buffer ----
#if GEMM_USE_TDM
      if (wave == 0) {
        unsigned long long ga = (unsigned long long)(const void*)(W + (size_t)k0n * N + bn0);
        u32x4 g0;
        g0[0] = 1u;
        g0[1] = lds_byte_offset(&sB[1 - buf][0]);
        g0[2] = (unsigned)ga;
        g0[3] = ((unsigned)(ga >> 32) & 0x01FFFFFFu) | 0x80000000u;
        __builtin_amdgcn_tensor_load_to_lds(g0, g1, gz, gz, gz8, 0);
      }
#else
      for (int i = tid; i < BK * BN; i += 256) {
        int kk = i >> 7, n = i & (BN - 1);
        int gk = k0n + kk, gn = bn0 + n;
        sB[1 - buf][i] = (gk < K && gn < N) ? W[(size_t)gk * N + gn] : (unsigned short)0;
      }
#endif
      // prefetch A tile s+2 into L2 (global_prefetch_b8)
      if (k0n + BK < K) {
        int pm = bm0 + (tid >> 1);
        if (pm < M) __builtin_prefetch(&A[(size_t)pm * K + k0n + BK + (tid & 1) * 16], 0, 1);
      }
      for (int i = tid; i < BM * BK; i += 256) {
        int m = i >> 5, kk = i & 31;
        int gm = bm0 + m, gk = k0n + kk;
        float v = (gm < M && gk < K) ? A[(size_t)gm * K + gk] : 0.0f;
        sA[1 - buf][m * BK + kk] = f2bf(v);
      }
    }

    // ---- compute on tile s ----
    Frag fa[2], fb[4];
#pragma unroll
    for (int mi = 0; mi < 2; ++mi) {
      int m  = wm * 32 + mi * 16 + mlane;
      int kb = lhalf * 8;
#pragma unroll
      for (int v = 0; v < 8; ++v) {
        int ks = (v < 4) ? (kb + 2 * v) : (16 + kb + 2 * (v - 4));
        fa[mi].u[v] = *(const unsigned int*)&sA[buf][m * BK + ks];
      }
    }
#pragma unroll
    for (int ni = 0; ni < 4; ++ni) {
      int n  = wn * 64 + ni * 16 + mlane;
      int kb = lhalf * 8;
#pragma unroll
      for (int v = 0; v < 8; ++v) {
        int ks = (v < 4) ? (kb + 2 * v) : (16 + kb + 2 * (v - 4));
        unsigned lo = sB[buf][ks * BN + n];
        unsigned hi = sB[buf][(ks + 1) * BN + n];
        fb[ni].u[v] = lo | (hi << 16);
      }
    }
#pragma unroll
    for (int mi = 0; mi < 2; ++mi)
#pragma unroll
      for (int ni = 0; ni < 4; ++ni)
        acc[mi][ni] = __builtin_amdgcn_wmma_f32_16x16x32_bf16(
            false, fa[mi].v, false, fb[ni].v, (short)0, acc[mi][ni], false, false);

#if GEMM_USE_TDM
    if (wave == 0) __builtin_amdgcn_s_wait_tensorcnt(0);
#endif
    __syncthreads();
  }

#pragma unroll
  for (int mi = 0; mi < 2; ++mi) {
#pragma unroll
    for (int ni = 0; ni < 4; ++ni) {
      int n = bn0 + wn * 64 + ni * 16 + mlane;
#pragma unroll
      for (int r = 0; r < 8; ++r) {
        int m = bm0 + wm * 32 + mi * 16 + r + lhalf * 8;
        if (m < M && n < N) {
          float v = acc[mi][ni][r] + (bias ? bias[n] : 0.0f);
          if (RELU) v = v > 0.0f ? v : 0.0f;
          C[(size_t)m * N + n] = v;
        }
      }
    }
  }
}

// ---------------- attention core: O = softmax(QK^T/8 + mask) V, one block per (b,h) ----------------
__global__ __launch_bounds__(256) void attn_kernel(
    const float* __restrict__ Q, const float* __restrict__ K,
    const float* __restrict__ V, float* __restrict__ O,
    const float* __restrict__ fmask,   // [B,Lk] key-padding mask (==0 -> blocked), or null
    const int*   __restrict__ imask,   // [B,DEPTH,LP,LP] transition mask, or null
    int step, int Lq, int Lk)
{
  __shared__ float sK[RREG * DHEAD];
  __shared__ float sV[RREG * DHEAD];
  const int b = blockIdx.x >> 3;
  const int h = blockIdx.x & 7;
  for (int i = threadIdx.x; i < Lk * DHEAD; i += 256) {
    int k = i >> 6, d = i & 63;
    size_t g = (size_t)(b * Lk + k) * HID + h * DHEAD + d;
    sK[i] = K[g];
    sV[i] = V[g];
  }
  __syncthreads();
  const int lane = threadIdx.x & 31;
  const int wave = threadIdx.x >> 5;
  for (int q = wave; q < Lq; q += 8) {
    size_t qg = (size_t)(b * Lq + q) * HID + h * DHEAD + 2 * lane;
    float q0 = Q[qg], q1 = Q[qg + 1];
    float sc0 = -1e30f, sc1 = -1e30f;     // scores for k=lane, k=lane+32
    for (int k = 0; k < Lk; ++k) {
      float p = q0 * sK[k * DHEAD + 2 * lane] + q1 * sK[k * DHEAD + 2 * lane + 1];
#pragma unroll
      for (int off = 16; off > 0; off >>= 1) p += __shfl_xor(p, off, 32);
      p *= 0.125f;  // 1/sqrt(64)
      bool blocked;
      if (imask) blocked = imask[(((size_t)b * DEPTH + step) * LP + q) * LP + k] == 0;
      else       blocked = fmask[(size_t)b * Lk + k] == 0.0f;
      if (blocked) p = -1e9f;
      if ((k & 31) == lane) { if (k < 32) sc0 = p; else sc1 = p; }
    }
    float mx = sc0 > sc1 ? sc0 : sc1;
#pragma unroll
    for (int off = 16; off > 0; off >>= 1) {
      float o = __shfl_xor(mx, off, 32);
      mx = o > mx ? o : mx;
    }
    float e0 = __expf(sc0 - mx);
    float e1 = __expf(sc1 - mx);
    if (lane >= Lk)      e0 = 0.0f;
    if (32 + lane >= Lk) e1 = 0.0f;
    float sum = e0 + e1;
#pragma unroll
    for (int off = 16; off > 0; off >>= 1) sum += __shfl_xor(sum, off, 32);
    float inv = 1.0f / sum;
    float p0 = e0 * inv, p1 = e1 * inv;
    float o0 = 0.0f, o1 = 0.0f;
    for (int k = 0; k < Lk; ++k) {
      float pk = __shfl((k < 32) ? p0 : p1, k & 31, 32);
      o0 += pk * sV[k * DHEAD + 2 * lane];
      o1 += pk * sV[k * DHEAD + 2 * lane + 1];
    }
    O[qg]     = o0;
    O[qg + 1] = o1;
  }
}

// ---------------- residual + layernorm (+ optional row mask), one wave per token ----------------
__global__ __launch_bounds__(128) void resln_kernel(
    float* __restrict__ out, const float* __restrict__ x, const float* __restrict__ r,
    const float* __restrict__ g, const float* __restrict__ bta,
    const float* __restrict__ rowmask, int ntok)
{
  const int lane = threadIdx.x & 31;
  const int wave = threadIdx.x >> 5;
  const int t = blockIdx.x * 4 + wave;
  if (t >= ntok) return;
  const float* px = x + (size_t)t * HID;
  const float* pr = r + (size_t)t * HID;
  float v[16];
  float s = 0.0f;
#pragma unroll
  for (int j = 0; j < 16; ++j) { int d = lane + j * 32; v[j] = px[d] + pr[d]; s += v[j]; }
#pragma unroll
  for (int off = 16; off > 0; off >>= 1) s += __shfl_xor(s, off, 32);
  float mean = s * (1.0f / HID);
  float var = 0.0f;
#pragma unroll
  for (int j = 0; j < 16; ++j) { float d = v[j] - mean; var += d * d; }
#pragma unroll
  for (int off = 16; off > 0; off >>= 1) var += __shfl_xor(var, off, 32);
  var *= (1.0f / HID);
  float inv = rsqrtf(var + 1e-5f);
  float msk = rowmask ? rowmask[t] : 1.0f;
  float* po = out + (size_t)t * HID;
#pragma unroll
  for (int j = 0; j < 16; ++j) {
    int d = lane + j * 32;
    po[d] = ((v[j] - mean) * inv * g[d] + bta[d]) * msk;
  }
}

// ---------------- scan gate: out = a ? new : old (a = activate_masks[b, step, lp]) ----------------
__global__ void gate_kernel(float* __restrict__ out, const float* __restrict__ nv,
                            const float* __restrict__ ov, const int* __restrict__ am,
                            int step, int ntok)
{
  int i = blockIdx.x * 256 + threadIdx.x;
  if (i >= ntok * HID) return;
  int t = i >> 9;
  int b = t >> 5, lp = t & 31;
  int a = am[((size_t)b * DEPTH + step) * LP + lp];
  out[i] = a ? nv[i] : ov[i];
}

// ---------------- question embedding + proj(300->512) + sinusoid pos ----------------
__global__ __launch_bounds__(256) void ques_embed_kernel(
    const int* __restrict__ ques, const float* __restrict__ emb,
    const float* __restrict__ W, const float* __restrict__ bias, float* __restrict__ out)
{
  __shared__ float e[EMBD];
  int t = blockIdx.x;                       // B*LQ tokens
  int tok = ques[t];
  for (int i = threadIdx.x; i < EMBD; i += 256) e[i] = emb[(size_t)tok * EMBD + i];
  __syncthreads();
  int pos = t & (LQ - 1);
  for (int h = threadIdx.x; h < HID; h += 256) {
    float s = bias[h];
    for (int d = 0; d < EMBD; ++d) s += e[d] * W[d * HID + h];
    float ang = (float)pos * __powf(10000.0f, -(float)(h & ~1) / (float)HID);
    s += (h & 1) ? __cosf(ang) : __sinf(ang);
    out[(size_t)t * HID + h] = s;
  }
}

// ---------------- program embedding + proj(300->64), sub-token -> dim slice ----------------
__global__ __launch_bounds__(64) void prog_embed_kernel(
    const int* __restrict__ prog, const float* __restrict__ emb,
    const float* __restrict__ W, const float* __restrict__ bias, float* __restrict__ out)
{
  __shared__ float e[EMBD];
  int t = blockIdx.x;                       // B*LP*SUB sub-tokens
  int tok = prog[t];
  for (int i = threadIdx.x; i < EMBD; i += 64) e[i] = emb[(size_t)tok * EMBD + i];
  __syncthreads();
  int h = threadIdx.x;                      // 0..63
  float s = bias[h];
  for (int d = 0; d < EMBD; ++d) s += e[d] * W[d * 64 + h];
  out[(size_t)t * 64 + h] = s;
}

// ---------------- vis += coord_proj(box) + pos_emb ----------------
__global__ void vis_add_kernel(float* __restrict__ vis, const float* __restrict__ box,
                               const float* __restrict__ Wc, const float* __restrict__ bc,
                               const float* __restrict__ pos)
{
  int t = blockIdx.x;                       // B*R
  int r = t % RREG;
  for (int h = threadIdx.x; h < HID; h += blockDim.x) {
    float s = bc[h];
#pragma unroll
    for (int d = 0; d < 6; ++d) s += box[(size_t)t * 6 + d] * Wc[d * HID + h];
    vis[(size_t)t * HID + h] += s + pos[(size_t)r * HID + h];
  }
}

// ---------------- lang gather + question mean -> concat buffer [B,1024] ----------------
__global__ void finalize_cat_kernel(const float* __restrict__ x, const float* __restrict__ enc,
                                    const int* __restrict__ index, float* __restrict__ cat)
{
  int b = blockIdx.x;
  int idx = index[b];
  for (int h = threadIdx.x; h < HID; h += blockDim.x) {
    float s = 0.0f;
    for (int q = 0; q < LQ; ++q) s += x[((size_t)b * LQ + q) * HID + h];
    cat[(size_t)b * 1024 + HID + h] = s * (1.0f / LQ);
    cat[(size_t)b * 1024 + h] = enc[((size_t)b * LP + idx) * HID + h];
  }
}

// ================= host side =================
struct Lin  { const float *w, *b; };
struct MHA  { Lin q, k, v, o; };
struct LNp  { const float *g, *b; };
struct FFNp { Lin fc, out; };
struct SAp  { MHA att;  FFNp ffn; LNp ln1, ln2; };
struct SGAp { MHA catt; FFNp ffn; LNp ln1, ln2, ln3; MHA satt; };
struct ShPp { MHA catt; FFNp ffn; LNp ln1, ln2; };
// bf16-weight versions
struct HLin { const unsigned short* w; const float* b; };
struct HMHA { HLin q, k, v, o; };
struct HFFN { HLin fc, out; };

static inline void launch_gemm(hipStream_t s, const float* A, const unsigned short* W,
                               const float* bias, float* C, int M, int N, int K, bool relu)
{
  dim3 grid((N + BN - 1) / BN, (M + BM - 1) / BM);
  if (relu) gemm_bias_kernel<true ><<<grid, 256, 0, s>>>(A, W, bias, C, M, N, K);
  else      gemm_bias_kernel<false><<<grid, 256, 0, s>>>(A, W, bias, C, M, N, K);
}

extern "C" void kernel_launch(void* const* d_in, const int* in_sizes, int n_in,
                              void* d_out, int out_size, void* d_ws, size_t ws_size,
                              hipStream_t stream) {
  (void)in_sizes; (void)n_in; (void)out_size; (void)ws_size;
  // ---- raw inputs ----
  const int*   ques    = (const int*)  d_in[0];
  const float* ques_m  = (const float*)d_in[1];
  const int*   program = (const int*)  d_in[2];
  const float* prog_m  = (const float*)d_in[3];
  const int*   trans_m = (const int*)  d_in[4];
  const int*   act_m   = (const int*)  d_in[5];
  const float* vis_ft  = (const float*)d_in[6];
  const float* box_ft  = (const float*)d_in[7];
  const float* vis_m   = (const float*)d_in[8];
  const int*   index   = (const int*)  d_in[9];
  // d_in[10] = depth (compile-time DEPTH)

  // ---- params: jax pytree flattened with sorted dict keys; each _lin -> (b, w), each _ln -> (b, g) ----
  int ip = 11;
  auto nf  = [&]() { return (const float*)d_in[ip++]; };
  auto lin = [&]() { Lin l; l.b = nf(); l.w = nf(); return l; };
  auto mha = [&]() { MHA m; m.k = lin(); m.o = lin(); m.q = lin(); m.v = lin(); return m; };
  auto lnp = [&]() { LNp l; l.b = nf(); l.g = nf(); return l; };
  auto ffp = [&]() { FFNp f; f.fc = lin(); f.out = lin(); return f; };

  Lin coord_proj = lin();
  const float* embedding = nf();
  Lin idx_pred = lin();
  SGAp mod; mod.catt = mha(); mod.ffn = ffp(); mod.ln1 = lnp(); mod.ln2 = lnp(); mod.ln3 = lnp(); mod.satt = mha();
  const float* pos_emb = nf();
  ShPp post[2];
  for (int i = 0; i < 2; ++i) { post[i].catt = mha(); post[i].ffn = ffp(); post[i].ln1 = lnp(); post[i].ln2 = lnp(); }
  Lin prog_proj = lin();
  Lin proj = lin();
  SAp qe[4];
  for (int i = 0; i < 4; ++i) { qe[i].att = mha(); qe[i].ffn = ffp(); qe[i].ln1 = lnp(); qe[i].ln2 = lnp(); }
  Lin ques_proj = lin();
  SGAp ve[4];
  for (int i = 0; i < 4; ++i) {
    ve[i].catt = mha(); ve[i].ffn = ffp(); ve[i].ln1 = lnp(); ve[i].ln2 = lnp(); ve[i].ln3 = lnp(); ve[i].satt = mha();
  }
  Lin vis_proj = lin();

  // ---- workspace layout ----
  const int NTQ = B_SZ * LQ;     // 8192
  const int NTV = B_SZ * RREG;   // 12288
  const int NTP = B_SZ * LP;     // 8192
  float* w = (float*)d_ws;
  size_t off = 0;
  auto alloc = [&](size_t n) { float* p = w + off; off += n; return p; };
  float* xq   = alloc((size_t)NTQ * HID);
  float* vis  = alloc((size_t)NTV * HID);
  float* encA = alloc((size_t)NTP * HID);
  float* encB = alloc((size_t)NTP * HID);
  float* T1   = alloc((size_t)NTV * HID);
  float* T2   = alloc((size_t)NTP * HID);
  float* Qb   = alloc((size_t)NTV * HID);
  float* Kb   = alloc((size_t)NTV * HID);
  float* Vb   = alloc((size_t)NTV * HID);
  float* AO   = alloc((size_t)NTV * HID);
  float* Kmod = alloc((size_t)NTV * HID);
  float* Vmod = alloc((size_t)NTV * HID);
  float* Hff  = alloc((size_t)NTV * FFDIM);
  float* cat  = alloc((size_t)B_SZ * 1024);
  float* outf = (float*)d_out;   // pre_logits [B,LP,48] then logits [B,1853]

  // ---- one-shot weight pre-conversion to bf16 (L2-resident afterwards) ----
  unsigned short* bf16p = (unsigned short*)(w + off);
  size_t boff = 0;
  auto cvtw = [&](const float* src, size_t n) -> const unsigned short* {
    unsigned short* d = bf16p + boff;
    boff += (n + 3) & ~(size_t)3;
    cvt_f32_bf16_kernel<<<((int)n + 255) / 256, 256, 0, stream>>>(src, d, (int)n);
    return d;
  };
  auto hlin = [&](const Lin& l, size_t k, size_t n) { HLin h; h.b = l.b; h.w = cvtw(l.w, k * n); return h; };
  auto hmha = [&](const MHA& m) {
    HMHA h; h.q = hlin(m.q, HID, HID); h.k = hlin(m.k, HID, HID);
    h.v = hlin(m.v, HID, HID); h.o = hlin(m.o, HID, HID); return h;
  };
  auto hffn = [&](const FFNp& f) {
    HFFN h; h.fc = hlin(f.fc, HID, FFDIM); h.out = hlin(f.out, FFDIM, HID); return h;
  };

  HMHA hqe_att[4]; HFFN hqe_ffn[4];
  for (int i = 0; i < 4; ++i) { hqe_att[i] = hmha(qe[i].att); hqe_ffn[i] = hffn(qe[i].ffn); }
  HMHA hve_satt[4], hve_catt[4]; HFFN hve_ffn[4];
  for (int i = 0; i < 4; ++i) {
    hve_satt[i] = hmha(ve[i].satt); hve_catt[i] = hmha(ve[i].catt); hve_ffn[i] = hffn(ve[i].ffn);
  }
  HMHA hmod_satt = hmha(mod.satt), hmod_catt = hmha(mod.catt);
  HFFN hmod_ffn  = hffn(mod.ffn);
  HMHA hpost_catt[2]; HFFN hpost_ffn[2];
  for (int i = 0; i < 2; ++i) { hpost_catt[i] = hmha(post[i].catt); hpost_ffn[i] = hffn(post[i].ffn); }
  HLin hvis_proj = hlin(vis_proj, 2048, HID);
  HLin hidx_pred = hlin(idx_pred, HID, RREG);
  HLin hproj     = hlin(proj, 1024, ANSV);

  auto run_ln = [&](float* o, const float* x, const float* r, const LNp& l,
                    const float* rowmask, int nt) {
    resln_kernel<<<(nt + 3) / 4, 128, 0, stream>>>(o, x, r, l.g, l.b, rowmask, nt);
  };
  auto run_attn = [&](const HMHA& m, const float* qin, const float* kin, int Lqq, int Lkk,
                      const float* fmask, const int* imask, int step, float* outb) {
    int ntq = B_SZ * Lqq, ntk = B_SZ * Lkk;
    launch_gemm(stream, qin, m.q.w, m.q.b, Qb, ntq, HID, HID, false);
    launch_gemm(stream, kin, m.k.w, m.k.b, Kb, ntk, HID, HID, false);
    launch_gemm(stream, kin, m.v.w, m.v.b, Vb, ntk, HID, HID, false);
    attn_kernel<<<B_SZ * NHEAD, 256, 0, stream>>>(Qb, Kb, Vb, AO, fmask, imask, step, Lqq, Lkk);
    launch_gemm(stream, AO, m.o.w, m.o.b, outb, ntq, HID, HID, false);
  };
  auto run_ffn = [&](const HFFN& f, const float* in, float* outb, int nt) {
    launch_gemm(stream, in, f.fc.w, f.fc.b, Hff, nt, FFDIM, HID, true);
    launch_gemm(stream, Hff, f.out.w, f.out.b, outb, nt, HID, FFDIM, false);
  };

  // 1) question embedding + proj + sinusoid
  ques_embed_kernel<<<NTQ, 256, 0, stream>>>(ques, embedding, ques_proj.w, ques_proj.b, xq);

  // 2) question self-attention encoder (4x SA)
  for (int l = 0; l < 4; ++l) {
    run_attn(hqe_att[l], xq, xq, LQ, LQ, ques_m, nullptr, 0, T1);
    run_ln(xq, xq, T1, qe[l].ln1, nullptr, NTQ);
    run_ffn(hqe_ffn[l], xq, T1, NTQ);
    run_ln(xq, xq, T1, qe[l].ln2, ques_m, NTQ);
  }

  // 3) visual features: vis_proj GEMM + coord proj + pos emb
  launch_gemm(stream, vis_ft, hvis_proj.w, hvis_proj.b, vis, NTV, HID, 2048, false);
  vis_add_kernel<<<NTV, 256, 0, stream>>>(vis, box_ft, coord_proj.w, coord_proj.b, pos_emb);

  // 4) visual encoder (4x SGA: self-att, cross-att to question, FFN)
  for (int l = 0; l < 4; ++l) {
    run_attn(hve_satt[l], vis, vis, RREG, RREG, vis_m, nullptr, 0, T1);
    run_ln(vis, vis, T1, ve[l].ln1, nullptr, NTV);
    run_attn(hve_catt[l], vis, xq, RREG, LQ, ques_m, nullptr, 0, T1);
    run_ln(vis, vis, T1, ve[l].ln2, nullptr, NTV);
    run_ffn(hve_ffn[l], vis, T1, NTV);
    run_ln(vis, vis, T1, ve[l].ln3, vis_m, NTV);
  }

  // 5) program embedding -> encA [B,LP,512]
  prog_embed_kernel<<<B_SZ * LP * SUBW, 64, 0, stream>>>(program, embedding,
                                                         prog_proj.w, prog_proj.b, encA);

  // 6) hoist module cross-attention K/V of vis (loop-invariant over depth)
  launch_gemm(stream, vis, hmod_catt.k.w, hmod_catt.k.b, Kmod, NTV, HID, HID, false);
  launch_gemm(stream, vis, hmod_catt.v.w, hmod_catt.v.b, Vmod, NTV, HID, HID, false);

  // 7) tree message-passing scan (8 depth steps, shared module weights)
  for (int d = 0; d < DEPTH; ++d) {
    // self-attention with transition mask
    launch_gemm(stream, encA, hmod_satt.q.w, hmod_satt.q.b, Qb, NTP, HID, HID, false);
    launch_gemm(stream, encA, hmod_satt.k.w, hmod_satt.k.b, Kb, NTP, HID, HID, false);
    launch_gemm(stream, encA, hmod_satt.v.w, hmod_satt.v.b, Vb, NTP, HID, HID, false);
    attn_kernel<<<B_SZ * NHEAD, 256, 0, stream>>>(Qb, Kb, Vb, AO, nullptr, trans_m, d, LP, LP);
    launch_gemm(stream, AO, hmod_satt.o.w, hmod_satt.o.b, T1, NTP, HID, HID, false);
    run_ln(T2, encA, T1, mod.ln1, nullptr, NTP);
    // cross-attention to vis (hoisted K/V)
    launch_gemm(stream, T2, hmod_catt.q.w, hmod_catt.q.b, Qb, NTP, HID, HID, false);
    attn_kernel<<<B_SZ * NHEAD, 256, 0, stream>>>(Qb, Kmod, Vmod, AO, vis_m, nullptr, 0, LP, RREG);
    launch_gemm(stream, AO, hmod_catt.o.w, hmod_catt.o.b, T1, NTP, HID, HID, false);
    run_ln(T2, T2, T1, mod.ln2, nullptr, NTP);
    // FFN
    run_ffn(hmod_ffn, T2, T1, NTP);
    run_ln(T2, T2, T1, mod.ln3, prog_m, NTP);
    // gate with activate mask, double-buffered
    gate_kernel<<<(NTP * HID + 255) / 256, 256, 0, stream>>>(encB, T2, encA, act_m, d, NTP);
    float* t = encA; encA = encB; encB = t;
  }

  // 8) post layers (2x shallow: cross-att to vis + FFN)
  for (int l = 0; l < 2; ++l) {
    run_attn(hpost_catt[l], encA, vis, LP, RREG, vis_m, nullptr, 0, T1);
    run_ln(encA, encA, T1, post[l].ln1, nullptr, NTP);
    run_ffn(hpost_ffn[l], encA, T1, NTP);
    run_ln(encA, encA, T1, post[l].ln2, prog_m, NTP);
  }

  // 9) pre_logits = idx_pred(enc) -> d_out[0 : B*LP*48]
  launch_gemm(stream, encA, hidx_pred.w, hidx_pred.b, outf, NTP, RREG, HID, false);

  // 10) logits = proj(concat(lang, mean(x))) -> d_out[B*LP*48 : ]
  finalize_cat_kernel<<<B_SZ, 256, 0, stream>>>(xq, encA, index, cat);
  launch_gemm(stream, cat, hproj.w, hproj.b, outf + (size_t)B_SZ * LP * RREG,
              B_SZ, ANSV, 1024, false);
}
